// Router_55748675502353
// MI455X (gfx1250) — compile-verified, matrix operand order checked by reading, not run on hardware
//
#include <hip/hip_runtime.h>
#include <hip/hip_bf16.h>
#include <math.h>

// ---------------------------------------------------------------------------
// Fused noisy-top-k router for MI455X (gfx1250, wave32, WMMA).
//   logits      = A @ gate_w^T  + gate_b     (A = mha_out, [M, D])
//   noiselogits = A @ noise_w^T + noise_b
//   noisy       = logits + stdnorm * softplus(noiselogits)
//   top-2 -> softmax over the 2 -> scatter into [M, E] (zeros elsewhere)
// Both GEMMs fused (N = 2E = 128) so the 256 MB activation tensor is streamed
// from HBM exactly once; bf16 WMMA w/ f32 accumulate keeps it on the
// 23.3 TB/s memory roof. 16 waves/block with a 2-way N-split keep per-wave
// VGPRs low enough that all B fragments stay live (pipelined dscnt waits)
// and occupancy at 4+ waves/SIMD.
// ---------------------------------------------------------------------------

typedef __attribute__((ext_vector_type(16))) __bf16 v16bf;
typedef __attribute__((ext_vector_type(2)))  __bf16 bf16x2;
typedef __attribute__((ext_vector_type(8)))  float  v8f;

#define TILE_M  128                // token rows per workgroup
#define TILE_K  32                 // K per WMMA step (bf16)
#define NCOMB   128                // 2*E fused output columns
#define NT_WAVE 4                  // N-tiles per wave (2-way split of 8)
#define BBUF    (NCOMB * TILE_K)   // bf16 elements per B ping-pong buffer (8KB)
#define BLOCK   512                // 16 wave32s

union BFrag { v16bf v; uint4 q[2]; };

__device__ __forceinline__ unsigned int pack2(float lo, float hi) {
  // native f32->bf16 (RNE): v_cvt_pk_bf16_f32
  union { bf16x2 v; unsigned int u; } c;
  c.v = (bf16x2){(__bf16)lo, (__bf16)hi};
  return c.u;
}

__global__ void __launch_bounds__(BLOCK)
router_wmma_kernel(const float* __restrict__ A,        // [M, D] activations
                   const float* __restrict__ gate_w,   // [E, D]
                   const float* __restrict__ gate_b,   // [E]
                   const float* __restrict__ noise_w,  // [E, D]
                   const float* __restrict__ noise_b,  // [E]
                   const float* __restrict__ stdn,     // [M, E]
                   float*       __restrict__ out_probs,// [M, E]
                   int*         __restrict__ out_idx,  // [M, 2]
                   int M, int D, int E)
{
  // 64 KB LDS: during the K loop the first 16 KB are two ping-pong bf16
  // B tiles; afterwards the whole buffer is the f32 logits scoreboard.
  __shared__ __align__(16) float s_mem[TILE_M * NCOMB];
  __bf16* s_b = (__bf16*)s_mem;

  const int tid   = threadIdx.x;
  const int wave  = tid >> 5;          // 0..15
  const int lane  = tid & 31;
  const int half  = lane >> 4;         // K-half select (WMMA 16-bit layout)
  const int l16   = lane & 15;
  const int rgrp  = wave >> 1;         // row group 0..7 (16 rows each)
  const int nhalf = wave & 1;          // which 4 of the 8 N-tiles

  const int m0 = blockIdx.x * TILE_M;  // workgroup token base
  const int mw = m0 + rgrp * 16;       // wave token base

  v8f acc[NT_WAVE] = {};               // 4 x (16x16 f32) = 16x64

  const float* Arow = A + (size_t)(mw + l16) * D;

  // B staging: 512 threads x 8 fp32 = 128x32 tile; one ds_store_b128 each
  const int brow = tid >> 2;                 // fused-weight row 0..127
  const int bcol = (tid & 3) * 8;            // 8-col chunk
  const float* Wrow = (brow < E) ? (gate_w  + (size_t)brow       * D)
                                 : (noise_w + (size_t)(brow - E) * D);
  const int sb_off = brow * TILE_K + bcol;

  // ---- prologue: stage weight K-tile 0 into buffer 0 ----------------------
  {
    const float4* src = (const float4*)(Wrow + bcol);
    float4 w0 = src[0], w1 = src[1];
    uint4 p;
    p.x = pack2(w0.x, w0.y);  p.y = pack2(w0.z, w0.w);
    p.z = pack2(w1.x, w1.y);  p.w = pack2(w1.z, w1.w);
    *(uint4*)(s_b + sb_off) = p;
  }
  __syncthreads();

  const int nsteps = D / TILE_K;
  for (int kt = 0; kt < nsteps; ++kt) {
    const int k0 = kt * TILE_K;
    const __bf16* cur = s_b + (kt & 1) * BBUF;
    __bf16*       nxt = s_b + ((kt + 1) & 1) * BBUF;
    const bool have_next = (kt + 1) < nsteps;

    // A fragment global loads (16-bit A 16x32 layout: lane half h covers
    // K = 8h+0..7 -> elems 0..7 and K = 16+8h+0..7 -> elems 8..15)
    const float* ap = Arow + k0 + 8 * half;
    float4 a0 = *(const float4*)(ap +  0);
    float4 a1 = *(const float4*)(ap +  4);
    float4 a2 = *(const float4*)(ap + 16);
    float4 a3 = *(const float4*)(ap + 20);
    __builtin_prefetch(ap + 2 * TILE_K, 0, 3);   // A tile for step kt+2

    // next weight K-tile global loads — latency hides under the WMMAs below
    float4 w0, w1;
    if (have_next) {
      const float4* src = (const float4*)(Wrow + k0 + TILE_K + bcol);
      w0 = src[0]; w1 = src[1];
    }

    // this wave's 4 B fragments into distinct registers (pipelined waits)
    BFrag bf[NT_WAVE];
    #pragma unroll
    for (int t = 0; t < NT_WAVE; ++t) {
      const int nt = nhalf * NT_WAVE + t;
      const __bf16* bp = cur + (nt * 16 + l16) * TILE_K + 8 * half;
      bf[t].q[0] = *(const uint4*)(bp);        // K = 8h+0..7
      bf[t].q[1] = *(const uint4*)(bp + 16);   // K = 16+8h+0..7
    }

    BFrag af;
    af.q[0].x = pack2(a0.x, a0.y);  af.q[0].y = pack2(a0.z, a0.w);
    af.q[0].z = pack2(a1.x, a1.y);  af.q[0].w = pack2(a1.z, a1.w);
    af.q[1].x = pack2(a2.x, a2.y);  af.q[1].y = pack2(a2.z, a2.w);
    af.q[1].z = pack2(a3.x, a3.y);  af.q[1].w = pack2(a3.z, a3.w);

    #pragma unroll
    for (int t = 0; t < NT_WAVE; ++t) {
      acc[t] = __builtin_amdgcn_wmma_f32_16x16x32_bf16(
          /*neg_a=*/false, af.v, /*neg_b=*/false, bf[t].v,
          /*c_mod=*/(short)0, acc[t], /*reuse_a=*/false, /*reuse_b=*/false);
    }

    // stage next weight tile into the other buffer
    if (have_next) {
      uint4 p;
      p.x = pack2(w0.x, w0.y);  p.y = pack2(w0.z, w0.w);
      p.z = pack2(w1.x, w1.y);  p.w = pack2(w1.z, w1.w);
      *(uint4*)(nxt + sb_off) = p;
    }
    __syncthreads();   // staging visible; cur readers done before reuse
  }

  // ---- dump accumulators to the LDS logits scoreboard ---------------------
  #pragma unroll
  for (int t = 0; t < NT_WAVE; ++t) {
    #pragma unroll
    for (int r = 0; r < 8; ++r) {
      // 32-bit C/D layout: VGPR r holds M=r (lanes 0-15) / M=r+8 (lanes 16-31)
      int mrow = rgrp * 16 + (half ? (r + 8) : r);
      int ncol = (nhalf * NT_WAVE + t) * 16 + l16;
      s_mem[mrow * NCOMB + ncol] = acc[t][r];
    }
  }
  __syncthreads();

  // ---- epilogue: one thread per token row --------------------------------
  if (tid < TILE_M) {
    const int   row = tid;
    const int   tok = m0 + row;
    const float* lg = s_mem + row * NCOMB;   // gate logits  [0,E)
    const float* nl = lg + E;                // noise logits [0,E)

    // zero-fill the output row first (scatter target)
    float4  z    = {0.f, 0.f, 0.f, 0.f};
    float4* orow = (float4*)(out_probs + (size_t)tok * E);
    for (int i = 0; i < E / 4; ++i) orow[i] = z;

    float v1 = -__builtin_inff(), v2 = -__builtin_inff();
    int   i1 = 0, i2 = 0;
    const float* sn = stdn + (size_t)tok * E;
    for (int e = 0; e < E; ++e) {
      float g  = lg[e] + gate_b[e];
      float x  = nl[e] + noise_b[e];
      float sp = fmaxf(x, 0.f) + log1pf(expf(-fabsf(x)));  // stable softplus
      float ny = fmaf(sn[e], sp, g);
      if (ny > v1)      { v2 = v1; i2 = i1; v1 = ny; i1 = e; }  // strict '>'
      else if (ny > v2) { v2 = ny; i2 = e; }                    // lower idx wins ties
    }
    float e2 = expf(v2 - v1);        // softmax over the two kept logits
    float p1 = 1.f / (1.f + e2);
    float p2 = e2 * p1;
    out_probs[(size_t)tok * E + i1] = p1;
    out_probs[(size_t)tok * E + i2] = p2;
    out_idx[tok * 2 + 0] = i1;
    out_idx[tok * 2 + 1] = i2;
  }
}

extern "C" void kernel_launch(void* const* d_in, const int* in_sizes, int n_in,
                              void* d_out, int out_size, void* d_ws, size_t ws_size,
                              hipStream_t stream) {
  const float* mha  = (const float*)d_in[0];   // [B,S,D]
  const float* gw   = (const float*)d_in[1];   // [E,D]
  const float* gb   = (const float*)d_in[2];   // [E]
  const float* nw   = (const float*)d_in[3];   // [E,D]
  const float* nb   = (const float*)d_in[4];   // [E]
  const float* stdn = (const float*)d_in[5];   // [B,S,E]
  // d_in[6] = topk (==2, hardwired in the epilogue)

  const int E = in_sizes[2];
  const int D = in_sizes[1] / E;
  const int M = in_sizes[0] / D;               // B*S tokens

  float* out_probs = (float*)d_out;                          // [M,E] f32
  int*   out_idx   = (int*)((float*)d_out + (size_t)M * E);  // [M,2] i32

  dim3 grid(M / TILE_M), block(BLOCK);
  hipLaunchKernelGGL(router_wmma_kernel, grid, block, 0, stream,
                     mha, gw, gb, nw, nb, stdn, out_probs, out_idx, M, D, E);
}
